// DotProductAttention_87754771792527
// MI455X (gfx1250) — matrix operand order repaired and checked by reading, not compiled, not run on hardware
//
#include <hip/hip_runtime.h>

typedef __attribute__((ext_vector_type(16))) __bf16 v16bf;
typedef __attribute__((ext_vector_type(8)))  __bf16 v8bf;
typedef __attribute__((ext_vector_type(2)))  __bf16 v2bf;
typedef __attribute__((ext_vector_type(8)))  float  v8f;

namespace {
constexpr int kB = 8, kS = 2048, kD = 64;
constexpr int kQTile = 16;   // query rows per wave
constexpr int kKTile = 32;   // keys per loop iteration (two 16-wide WMMA tiles)
constexpr int kWaves = 8;    // 256 threads -> 128 queries per block
constexpr int kNIter = kS / kKTile;
constexpr float kNeg   = -1e9f;
constexpr float kScale = 0.125f;                  // 1/sqrt(64)
constexpr float kLog2e = 1.44269504088896340736f;
}

static __device__ __forceinline__ __bf16 f2bf(float f) { return (__bf16)f; }

static __device__ __forceinline__ v8bf cvt8v(float4 a, float4 b) {
  v8bf r;
  r[0] = f2bf(a.x); r[1] = f2bf(a.y); r[2] = f2bf(a.z); r[3] = f2bf(a.w);
  r[4] = f2bf(b.x); r[5] = f2bf(b.y); r[6] = f2bf(b.z); r[7] = f2bf(b.w);
  return r;
}

// nibble-select word for "lane -> lane ^ M" within a 16-lane row
static constexpr unsigned selw(int m, int base) {
  unsigned v = 0;
  for (int i = 0; i < 8; ++i)
    v |= (unsigned)(((base + i) ^ m) & 0xF) << (4 * i);
  return v;
}

// xor-shuffle across a 16-lane row via v_permlane16_b32 (pure VALU, no LDS)
template <int M>
static __device__ __forceinline__ float permxor16(float x) {
#if __has_builtin(__builtin_amdgcn_permlane16)
  int i = __float_as_int(x);
  i = __builtin_amdgcn_permlane16(i, i, selw(M, 0), selw(M, 8), false, false);
  return __int_as_float(i);
#else
  return __shfl_xor(x, M, 32);
#endif
}

static __device__ __forceinline__ float rowmax16(float t) {
  t = fmaxf(t, permxor16<1>(t));
  t = fmaxf(t, permxor16<2>(t));
  t = fmaxf(t, permxor16<4>(t));
  t = fmaxf(t, permxor16<8>(t));
  return t;
}
static __device__ __forceinline__ float rowsum16(float t) {
  t += permxor16<1>(t);
  t += permxor16<2>(t);
  t += permxor16<4>(t);
  t += permxor16<8>(t);
  return t;
}

static __device__ __forceinline__ float fast_exp2(float x) {
#if __has_builtin(__builtin_amdgcn_exp2f)
  return __builtin_amdgcn_exp2f(x);   // raw v_exp_f32
#else
  return exp2f(x);
#endif
}

union ABFrag { v16bf v; v8bf h[2]; };

// registers holding one in-flight staged tile (next iteration's K/V rows)
struct Stage {
  float4 ka, kb2;   // this thread's K row chunk (8 f32)
  float4 va, vb;    // this thread's two V row chunks (d-group of 4, keys k0,k0+1)
  float  m0, m1;    // mask_v for keys k0, k0+1
};

__global__ __launch_bounds__(256, 2)
void fa_fwd(const float* __restrict__ Q, const float* __restrict__ K,
            const float* __restrict__ V, const float* __restrict__ MQ,
            const float* __restrict__ MK, const float* __restrict__ MV,
            float* __restrict__ O)
{
  // ping-pong staged tiles + per-wave P scratch
  __shared__ __align__(16) __bf16 sK [2][kKTile * kD];            // [key][d]
  __shared__ __align__(16) __bf16 sVt[2][kD * kKTile];            // [d][key], pre-masked
  __shared__ __align__(16) __bf16 sP [kWaves][kQTile * kKTile];

  const int tid  = threadIdx.x;
  const int wave = tid >> 5;
  const int lane = tid & 31;
  const int n    = lane & 15;   // column index within 16
  const int hh   = lane >> 4;   // half-wave select

  const int b  = blockIdx.x >> 4;                                 // 16 blocks/batch
  const int q0 = (blockIdx.x & 15) * (kQTile * kWaves) + wave * kQTile;

  const float* Qb = Q + (size_t)b * kS * kD;
  const float* Kb = K + (size_t)b * kS * kD;
  const float* Vb = V + (size_t)b * kS * kD;
  const float* mq = MQ + (size_t)b * kS;
  const float* mk = MK + (size_t)b * kS;
  const float* mv = MV + (size_t)b * kS;

  // staging geometry for this thread
  const int kr  = tid >> 3, kc0 = (tid & 7) * 8;   // K: row, col8
  const int vp  = tid >> 4, vdg = tid & 15;        // V: key-pair, d-group

  // ---- Q A-fragments (16x32 bf16 each, contiguous k = 16*hh + s order) ----
  ABFrag aQ[2];
  {
    const float* qrow = Qb + (size_t)(q0 + n) * kD;
#pragma unroll
    for (int w = 0; w < 2; ++w) {
      aQ[w].h[0] = cvt8v(*(const float4*)(qrow + w * 32 + hh * 16),
                         *(const float4*)(qrow + w * 32 + hh * 16 + 4));
      aQ[w].h[1] = cvt8v(*(const float4*)(qrow + w * 32 + hh * 16 + 8),
                         *(const float4*)(qrow + w * 32 + hh * 16 + 12));
    }
  }

  float qn[8];   // mask_q * 1e9 (score = cc*scale + NEG + qn*mk)
#pragma unroll
  for (int j = 0; j < 8; ++j) qn[j] = mq[q0 + j + 8 * hh] * (-kNeg);

  float mrow[8], srow[8];
  v8f acc[4];
#pragma unroll
  for (int j = 0; j < 8; ++j) { mrow[j] = -3.0e38f; srow[j] = 0.f; }
#pragma unroll
  for (int t = 0; t < 4; ++t)
#pragma unroll
    for (int j = 0; j < 8; ++j) acc[t][j] = 0.f;

  // ---- stage helpers (lambdas keep thread-geometry captured) ----
  auto stage_load = [&](int kb0) -> Stage {
    Stage s;
    const float* krow = Kb + (size_t)(kb0 + kr) * kD + kc0;
    s.ka  = *(const float4*)krow;
    s.kb2 = *(const float4*)(krow + 4);
    int k0 = kb0 + vp * 2;
    s.m0 = mv[k0]; s.m1 = mv[k0 + 1];
    s.va = *(const float4*)(Vb + (size_t)k0 * kD + vdg * 4);
    s.vb = *(const float4*)(Vb + (size_t)(k0 + 1) * kD + vdg * 4);
    return s;
  };
  auto stage_store = [&](const Stage& s, int buf) {
    *(v8bf*)&sK[buf][kr * kD + kc0] = cvt8v(s.ka, s.kb2);
    float e0[4] = {s.va.x, s.va.y, s.va.z, s.va.w};
    float e1[4] = {s.vb.x, s.vb.y, s.vb.z, s.vb.w};
#pragma unroll
    for (int i = 0; i < 4; ++i) {
      v2bf pr;
      pr[0] = f2bf(e0[i] * s.m0);
      pr[1] = f2bf(e1[i] * s.m1);
      *(v2bf*)&sVt[buf][(vdg * 4 + i) * kKTile + vp * 2] = pr;
    }
  };

  // ---- prologue: stage tile 0 ----
  stage_store(stage_load(0), 0);
  __syncthreads();

  // ---- 2-stage pipelined main loop: one barrier per tile ----
  for (int it = 0; it < kNIter; ++it) {
    const int cur = it & 1;
    const int kb0 = it * kKTile;

    Stage nxt;
    if (it + 1 < kNIter) nxt = stage_load(kb0 + kKTile);   // loads fly over compute

    // ---- preload all K B-fragments, then issue QK^T WMMAs back-to-back ----
    ABFrag bK[2][2];
#pragma unroll
    for (int ct = 0; ct < 2; ++ct)
#pragma unroll
      for (int w = 0; w < 2; ++w) {
        const __bf16* krp = &sK[cur][(ct * 16 + n) * kD + w * 32 + hh * 16];
        bK[ct][w].h[0] = *(const v8bf*)krp;
        bK[ct][w].h[1] = *(const v8bf*)(krp + 8);
      }
    v8f cc[2];
#pragma unroll
    for (int ct = 0; ct < 2; ++ct) {
      v8f c = {0.f, 0.f, 0.f, 0.f, 0.f, 0.f, 0.f, 0.f};
      c = __builtin_amdgcn_wmma_f32_16x16x32_bf16(
              false, aQ[0].v, false, bK[ct][0].v, (short)0, c, false, false);
      c = __builtin_amdgcn_wmma_f32_16x16x32_bf16(
              false, aQ[1].v, false, bK[ct][1].v, (short)0, c, false, false);
      cc[ct] = c;
    }

    // ---- scale + additive padding mask + online softmax (pure VALU) ----
    float mk0 = mk[kb0 + n];
    float mk1 = mk[kb0 + 16 + n];
    float p0[8], p1[8];
#pragma unroll
    for (int j = 0; j < 8; ++j) {
      float s0 = fmaf(cc[0][j], kScale, fmaf(qn[j], mk0, kNeg));
      float s1 = fmaf(cc[1][j], kScale, fmaf(qn[j], mk1, kNeg));
      float t  = rowmax16(fmaxf(s0, s1));
      float mn = fmaxf(mrow[j], t);
      float cf = fast_exp2((mrow[j] - mn) * kLog2e);
      float e0 = fast_exp2((s0 - mn) * kLog2e);
      float e1 = fast_exp2((s1 - mn) * kLog2e);
      float rs = rowsum16(e0 + e1);
      srow[j] = srow[j] * cf + rs;
      mrow[j] = mn;
      p0[j] = e0; p1[j] = e1;
      acc[0][j] *= cf; acc[1][j] *= cf; acc[2][j] *= cf; acc[3][j] *= cf;
    }

    // ---- C-layout -> A-layout for P via per-wave LDS scratch ----
    __bf16* pb = sP[wave];
#pragma unroll
    for (int j = 0; j < 8; ++j) {
      int row = j + 8 * hh;
      v2bf pr; pr[0] = f2bf(p0[j]); pr[1] = f2bf(p1[j]);
      pb[row * kKTile + n]      = pr[0];
      pb[row * kKTile + 16 + n] = pr[1];
    }
    ABFrag aP;   // same-wave LDS RAW ordered by DScnt
    aP.h[0] = *(const v8bf*)&pb[n * kKTile + hh * 16];
    aP.h[1] = *(const v8bf*)&pb[n * kKTile + hh * 16 + 8];

    // ---- preload all V B-fragments, then 4 independent PV WMMAs ----
    ABFrag bV[4];
#pragma unroll
    for (int dt = 0; dt < 4; ++dt) {
      const __bf16* vr = &sVt[cur][(dt * 16 + n) * kKTile + hh * 16];
      bV[dt].h[0] = *(const v8bf*)vr;
      bV[dt].h[1] = *(const v8bf*)(vr + 8);
    }
#pragma unroll
    for (int dt = 0; dt < 4; ++dt)
      acc[dt] = __builtin_amdgcn_wmma_f32_16x16x32_bf16(
                    false, aP.v, false, bV[dt].v, (short)0, acc[dt], false, false);

    // ---- commit next tile to the other LDS buffer, single barrier ----
    if (it + 1 < kNIter) stage_store(nxt, cur ^ 1);
    __syncthreads();
  }

  // ---- epilogue: normalize and store fp32 output ----
#pragma unroll
  for (int j = 0; j < 8; ++j) {
    float inv = 1.f / srow[j];
    int row = j + 8 * hh;
    float* orow = O + ((size_t)b * kS + q0 + row) * kD;
    orow[ 0 + n] = acc[0][j] * inv;
    orow[16 + n] = acc[1][j] * inv;
    orow[32 + n] = acc[2][j] * inv;
    orow[48 + n] = acc[3][j] * inv;
  }
}

extern "C" void kernel_launch(void* const* d_in, const int* in_sizes, int n_in,
                              void* d_out, int out_size, void* d_ws, size_t ws_size,
                              hipStream_t stream) {
  (void)in_sizes; (void)n_in; (void)out_size; (void)d_ws; (void)ws_size;
  const float* Q  = (const float*)d_in[0];
  const float* K  = (const float*)d_in[1];
  const float* V  = (const float*)d_in[2];
  const float* MQ = (const float*)d_in[3];
  const float* MK = (const float*)d_in[4];
  const float* MV = (const float*)d_in[5];
  float* O = (float*)d_out;

  dim3 grid(kB * (kS / (kQTile * kWaves)));   // 8 * 16 = 128 blocks
  dim3 block(256);                            // 8 waves (wave32)
  hipLaunchKernelGGL(fa_fwd, grid, block, 0, stream, Q, K, V, MQ, MK, MV, O);
}
